// Encoder_71287867179628
// MI455X (gfx1250) — matrix operand order repaired and checked by reading, not compiled, not run on hardware
//
#include <hip/hip_runtime.h>
#include <hip/hip_bf16.h>

typedef __bf16 bf16;
typedef __attribute__((ext_vector_type(16))) __bf16 bf16x16;
typedef __attribute__((ext_vector_type(8)))  float  f32x8;

constexpr int Bn = 64, Tn = 1024, En = 256, Hn = 512, G4 = 2048, Vn = 32000;
constexpr int BT = Bn * Tn;
constexpr long long YS = (long long)Bn * Tn * Hn;   // ys element count

union Frag16 { bf16x16 v; uint4 q[2]; };

// A-fragment (16x32 bf16): lane<16 -> K=[k0..k0+7] in v0-3, K=[k0+16..k0+23] in v4-7
//                          lane>=16 -> +8 on both chunks
__device__ __forceinline__ bf16x16 ldA(const bf16* row, int k0, bool hi) {
  Frag16 f;
  const int o = k0 + (hi ? 8 : 0);
  f.q[0] = *reinterpret_cast<const uint4*>(row + o);
  f.q[1] = *reinterpret_cast<const uint4*>(row + o + 16);
  return f.v;
}
// B-fragment (32x16 bf16) from transposed weights (row = N, contiguous K):
// lane<16 -> K=[k0..k0+7],[k0+16..k0+23]; lane>=16 -> +8 on both chunks
__device__ __forceinline__ bf16x16 ldB(const bf16* row, int k0, bool hi) {
  Frag16 f;
  const int o = k0 + (hi ? 8 : 0);
  f.q[0] = *reinterpret_cast<const uint4*>(row + o);
  f.q[1] = *reinterpret_cast<const uint4*>(row + o + 16);
  return f.v;
}

__device__ __forceinline__ f32x8 wmma_bf16(bf16x16 a, bf16x16 b, f32x8 c) {
  return __builtin_amdgcn_wmma_f32_16x16x32_bf16(false, a, false, b, (short)0, c, false, false);
}

__device__ __forceinline__ float sigm(float v) { return 1.0f / (1.0f + __expf(-v)); }

// ---------------- prep kernels ----------------
__global__ void cvt_bf16(const float* __restrict__ s, bf16* __restrict__ d, int n) {
  int i = blockIdx.x * blockDim.x + threadIdx.x;
  int stride = gridDim.x * blockDim.x;
  for (; i < n; i += stride) d[i] = (bf16)s[i];
}

// src [K][N] row-major f32 -> dst [N][K] bf16
__global__ void transpose_bf16(const float* __restrict__ s, bf16* __restrict__ d, int K, int N) {
  int i = blockIdx.x * blockDim.x + threadIdx.x;
  if (i < K * N) {
    int k = i / N, n = i % N;
    d[(size_t)n * K + k] = (bf16)s[i];
  }
}

__global__ void init_state(bf16* __restrict__ hbf, unsigned* __restrict__ ctr) {
  int i = blockIdx.x * blockDim.x + threadIdx.x;
  if (i < 2 * Bn * Hn) hbf[i] = (bf16)0.0f;
  if (i == 0) *ctr = 0u;
}

// ---------------- input projection: zx[b*T+t][0..2047] = emb[x] @ W + bias ----------------
// Each wave owns one n-group (64 columns = 4 n-tiles) with Wt fragments resident in
// registers (4 tiles x 8 k-chunks), then grid-strides over m-tiles.
__global__ void __launch_bounds__(256) zx_gemm(const bf16* __restrict__ emb_bf,
                                               const int* __restrict__ x,
                                               const bf16* __restrict__ Wt,   // [2048][256] bf16
                                               const float* __restrict__ bias,
                                               float* __restrict__ zx) {
  const int lane = threadIdx.x & 31;
  const bool hi = lane >= 16;
  const int ncol = lane & 15;
  const int wave  = (blockIdx.x * blockDim.x + threadIdx.x) >> 5;
  const int nwave = (gridDim.x * blockDim.x) >> 5;
  const int ng = wave & 31;            // n-group: 32 groups of 64 columns
  const int n0 = ng * 64;
  const int mstart  = wave >> 5;       // m-tile start
  const int mstride = nwave >> 5;

  // Preload B (Wt) fragments for this n-group: resident across all m-tiles.
  bf16x16 bfrag[4][8];
  float bv[4];
#pragma unroll
  for (int s = 0; s < 4; ++s) {
    const int n = n0 + s * 16 + ncol;
    const bf16* Brow = Wt + (size_t)n * En;
    bv[s] = bias[n];
#pragma unroll
    for (int kc = 0; kc < 8; ++kc)
      bfrag[s][kc] = ldB(Brow, kc * 32, hi);
  }

  for (int mt = mstart; mt < BT / 16; mt += mstride) {
    const int r = mt * 16 + ncol;               // global row = b*T+t
    const bf16* Arow = emb_bf + (size_t)x[r] * En;
    // Load full A row fragments up front (one clause), then pure WMMA chain.
    bf16x16 afrag[8];
#pragma unroll
    for (int kc = 0; kc < 8; ++kc)
      afrag[kc] = ldA(Arow, kc * 32, hi);

    f32x8 acc[4];
#pragma unroll
    for (int s = 0; s < 4; ++s)
#pragma unroll
      for (int v = 0; v < 8; ++v) acc[s][v] = bv[s];

#pragma unroll
    for (int kc = 0; kc < 8; ++kc)
#pragma unroll
      for (int s = 0; s < 4; ++s)
        acc[s] = wmma_bf16(afrag[kc], bfrag[s][kc], acc[s]);

#pragma unroll
    for (int s = 0; s < 4; ++s) {
      const int n = n0 + s * 16 + ncol;
#pragma unroll
      for (int v = 0; v < 8; ++v) {
        const int row = mt * 16 + v + (hi ? 8 : 0);
        zx[(size_t)row * G4 + n] = acc[s][v];
      }
    }
  }
}

// ---------------- persistent sequential LSTM ----------------
// Grid MUST be 32 blocks x 128 threads = 128 waves = 1 wave per SIMD32.
// wave w: m = w>>5 (batch tile), j = w&31 (hidden tile). Each wave owns a
// 16x16 output tile for all 4 gates; h,c live in registers for the whole
// sequence, ALL U fragments (4 gates x 16 k-chunks = 512 VGPRs) are pinned
// in registers. Per step: prefetch the full A row (8 x 128B lines per lane)
// into the near cache with global_prefetch, then run the rolling
// load -> 4-WMMA chunks; zx/mask loads retire underneath the WMMAs.
__global__ void __launch_bounds__(128) lstm_seq(const float* __restrict__ zx,
                                                const int* __restrict__ x,
                                                const bf16* __restrict__ Ut,   // [2048][512] bf16
                                                bf16* __restrict__ hbf,        // double buffer 2*[64][512]
                                                float* __restrict__ out,
                                                unsigned* __restrict__ ctr,
                                                int reverse, int accumulate) {
  const int lane = threadIdx.x & 31;
  const bool hi = lane >= 16;
  const int w = blockIdx.x * 4 + (threadIdx.x >> 5);
  const int m = w >> 5;                 // 0..3
  const int j = w & 31;                 // 0..31
  const int ncol = lane & 15;
  const int col = j * 16 + ncol;        // hidden column 0..511
  const int arow = m * 16 + ncol;       // batch row for A fragment

  // Pin all recurrent-weight fragments in registers for the whole sequence.
  bf16x16 bfrag[4][16];
  {
    const bf16* Bg0 = Ut + (size_t)(col) * Hn;
    const bf16* Bg1 = Ut + (size_t)(512 + col) * Hn;
    const bf16* Bg2 = Ut + (size_t)(1024 + col) * Hn;
    const bf16* Bg3 = Ut + (size_t)(1536 + col) * Hn;
#pragma unroll
    for (int kc = 0; kc < 16; ++kc) {
      bfrag[0][kc] = ldB(Bg0, kc * 32, hi);
      bfrag[1][kc] = ldB(Bg1, kc * 32, hi);
      bfrag[2][kc] = ldB(Bg2, kc * 32, hi);
      bfrag[3][kc] = ldB(Bg3, kc * 32, hi);
    }
  }

  float hr[8], cr[8];
#pragma unroll
  for (int v = 0; v < 8; ++v) { hr[v] = 0.0f; cr[v] = 0.0f; }
  const unsigned nb = gridDim.x;

  for (int t = 0; t < Tn; ++t) {
    const int te = reverse ? (Tn - 1 - t) : t;
    const bf16* hin = hbf + ((t & 1) ? Bn * Hn : 0);
    bf16* hout      = hbf + ((t & 1) ? 0 : Bn * Hn);

    // 1) Pull this step's entire A row into the near cache (no VGPR cost):
    //    each lane's row slice is 1KiB contiguous = 8 cache lines.
    const bf16* Arow = hin + arow * Hn;
#pragma unroll
    for (int pf = 0; pf < 8; ++pf)
      __builtin_prefetch((const void*)(Arow + pf * 64), 0, 3);

    // 2) zx / mask loads — retire under the WMMA chain, used in epilogue.
    float zq0[8], zq1[8], zq2[8], zq3[8];
    bool  mq[8];
#pragma unroll
    for (int v = 0; v < 8; ++v) {
      const int brow = m * 16 + v + (hi ? 8 : 0);
      const size_t zb = ((size_t)brow * Tn + te) * (size_t)G4 + col;
      zq0[v] = zx[zb];
      zq1[v] = zx[zb + 512];
      zq2[v] = zx[zb + 1024];
      zq3[v] = zx[zb + 1536];
      mq[v]  = x[brow * Tn + te] != 0;
    }

    // 3) Rolling A-chunk load + 4-WMMA groups (B resident in registers).
    f32x8 a0 = {}, a1 = {}, a2 = {}, a3 = {};
#pragma unroll
    for (int kc = 0; kc < 16; ++kc) {
      bf16x16 a = ldA(Arow, kc * 32, hi);
      a0 = wmma_bf16(a, bfrag[0][kc], a0);
      a1 = wmma_bf16(a, bfrag[1][kc], a1);
      a2 = wmma_bf16(a, bfrag[2][kc], a2);
      a3 = wmma_bf16(a, bfrag[3][kc], a3);
    }

    // 4) Gate math + state update + publish.
#pragma unroll
    for (int v = 0; v < 8; ++v) {
      const int brow = m * 16 + v + (hi ? 8 : 0);
      const float zi = a0[v] + zq0[v];
      const float zf = a1[v] + zq1[v];
      const float zg = a2[v] + zq2[v];
      const float zo = a3[v] + zq3[v];
      const float ig = sigm(zi), fg = sigm(zf), og = sigm(zo);
      const float gg = tanhf(zg);
      const float cn = fg * cr[v] + ig * gg;
      const float hn = og * tanhf(cn);
      const float hc = mq[v] ? hn : hr[v];
      const float cc = mq[v] ? cn : cr[v];
      hr[v] = hc; cr[v] = cc;
      hout[brow * Hn + col] = (bf16)hc;
      const size_t oy = ((size_t)brow * Tn + te) * (size_t)Hn + col;
      if (accumulate) out[oy] += hc; else out[oy] = hc;
    }

    // grid-wide barrier: publish hout, wait for everyone
    __syncthreads();
    if (threadIdx.x == 0) {
      __threadfence();
      __hip_atomic_fetch_add(ctr, 1u, __ATOMIC_RELEASE, __HIP_MEMORY_SCOPE_AGENT);
      const unsigned tgt = (unsigned)(t + 1) * nb;
      while (__hip_atomic_load(ctr, __ATOMIC_ACQUIRE, __HIP_MEMORY_SCOPE_AGENT) < tgt)
        __builtin_amdgcn_s_sleep(1);
    }
    __syncthreads();
  }

  // final hT / cT
#pragma unroll
  for (int v = 0; v < 8; ++v) {
    const int brow = m * 16 + v + (hi ? 8 : 0);
    const size_t oh = (size_t)YS + (size_t)brow * Hn + col;
    if (accumulate) { out[oh] += hr[v]; out[oh + Bn * Hn] += cr[v]; }
    else            { out[oh]  = hr[v]; out[oh + Bn * Hn]  = cr[v]; }
  }
}

extern "C" void kernel_launch(void* const* d_in, const int* in_sizes, int n_in,
                              void* d_out, int out_size, void* d_ws, size_t ws_size,
                              hipStream_t stream) {
  (void)in_sizes; (void)n_in; (void)out_size; (void)ws_size;
  const int*   x    = (const int*)d_in[0];
  const float* emb  = (const float*)d_in[1];
  const float* Wd[2] = {(const float*)d_in[2], (const float*)d_in[5]};
  const float* Ud[2] = {(const float*)d_in[3], (const float*)d_in[6]};
  const float* bd[2] = {(const float*)d_in[4], (const float*)d_in[7]};
  float* out = (float*)d_out;

  char* ws = (char*)d_ws;
  size_t off = 0;
  auto take = [&](size_t bytes) -> char* {
    char* p = ws + off;
    off += (bytes + 255) & ~(size_t)255;
    return p;
  };
  bf16*     emb_bf = (bf16*)take((size_t)Vn * En * sizeof(bf16));     // 16 MiB
  bf16*     Wt     = (bf16*)take((size_t)G4 * En * sizeof(bf16));     // 1 MiB
  bf16*     Ut     = (bf16*)take((size_t)G4 * Hn * sizeof(bf16));     // 2 MiB
  float*    zx     = (float*)take((size_t)BT * G4 * sizeof(float));   // 512 MiB
  bf16*     hbf    = (bf16*)take((size_t)2 * Bn * Hn * sizeof(bf16)); // 128 KiB
  unsigned* ctr    = (unsigned*)take(256);

  cvt_bf16<<<4096, 256, 0, stream>>>(emb, emb_bf, Vn * En);
  for (int d = 0; d < 2; ++d) {
    transpose_bf16<<<(En * G4 + 255) / 256, 256, 0, stream>>>(Wd[d], Wt, En, G4);
    transpose_bf16<<<(Hn * G4 + 255) / 256, 256, 0, stream>>>(Ud[d], Ut, Hn, G4);
    init_state<<<(2 * Bn * Hn + 255) / 256, 256, 0, stream>>>(hbf, ctr);
    zx_gemm<<<1024, 256, 0, stream>>>(emb_bf, x, Wt, bd[d], zx);
    lstm_seq<<<32, 128, 0, stream>>>(zx, x, Ut, hbf, out, ctr, /*reverse=*/d, /*accumulate=*/d);
  }
}
